// TextEncoder_24859270709810
// MI455X (gfx1250) — matrix operand order repaired and checked by reading, not compiled, not run on hardware
//
#include <hip/hip_runtime.h>
#include <hip/hip_bf16.h>
#include <math.h>

typedef __attribute__((ext_vector_type(16))) _Float16 v16h;
typedef __attribute__((ext_vector_type(8)))  _Float16 v8h;
typedef __attribute__((ext_vector_type(4)))  _Float16 v4h;
typedef __attribute__((ext_vector_type(8)))  float    v8f;
typedef __attribute__((ext_vector_type(4)))  int      v4i;

#define LN_EPS 1e-5f

#if defined(__gfx1250__) && __has_builtin(__builtin_amdgcn_global_load_async_to_lds_b128)
#define HAVE_ASYNC_LDS 1
#else
#define HAVE_ASYNC_LDS 0
#endif

// Copy 16 bytes global -> LDS.  Async (no VGPR round trip, ASYNCcnt-tracked)
// when the toolchain exposes the gfx1250 builtin, else a plain batched copy.
// Builtin signature (from probe): (int4 AS1*, int4 AS3*, imm offset, imm cpol).
__device__ __forceinline__ void async_copy16(const _Float16* g, _Float16* l) {
#if HAVE_ASYNC_LDS
  __builtin_amdgcn_global_load_async_to_lds_b128(
      (__attribute__((address_space(1))) v4i*)(g),
      (__attribute__((address_space(3))) v4i*)(l), 0, 0);
#else
  *(v8h*)l = *(const v8h*)g;
#endif
}

__device__ __forceinline__ void wait_async_all() {
#if HAVE_ASYNC_LDS
#if __has_builtin(__builtin_amdgcn_s_wait_asynccnt)
  __builtin_amdgcn_s_wait_asynccnt(0);
#else
  asm volatile("s_wait_asynccnt 0x0" ::: "memory");
#endif
#endif
}

__device__ __forceinline__ v8f wmma_f16(v16h a, v16h b, v8f c) {
  return __builtin_amdgcn_wmma_f32_16x16x32_f16(
      false, a, false, b, (short)0, c, false, false);
}

// A-fragment (16x32 f16) from a row-major LDS tile (row stride ld halves).
__device__ __forceinline__ v16h frag_a(const _Float16* base, int ld) {
  const int lane = threadIdx.x & 31;
  const _Float16* p = base + (lane & 15) * ld + ((lane >> 4) << 3);
  v8h lo = *(const v8h*)(p);
  v8h hi = *(const v8h*)(p + 16);
  v16h r;
#pragma unroll
  for (int i = 0; i < 8; ++i) { r[i] = lo[i]; r[i + 8] = hi[i]; }
  return r;
}

// B-fragment (32x16 f16) from an LDS tile stored [n][k] (n stride ld halves).
__device__ __forceinline__ v16h frag_b(const _Float16* base, int ld) {
  const int lane = threadIdx.x & 31;
  const _Float16* p = base + (lane & 15) * ld + ((lane >> 4) << 3);
  v8h lo = *(const v8h*)(p);
  v8h hi = *(const v8h*)(p + 16);
  v16h r;
#pragma unroll
  for (int i = 0; i < 8; ++i) { r[i] = lo[i]; r[i + 8] = hi[i]; }
  return r;
}

__device__ __forceinline__ float gelu_exact(float x) {
  return 0.5f * x * (1.0f + erff(x * 0.70710678118654752f));
}

// ---------------------------------------------------------------------------
// Pre-pass: W[K][N] f32  ->  WT[N][K] f16 (transposed), 64x64 tiles via LDS.
// Makes every GEMM B-tile a raw byte copy (async-LDS friendly).
// ---------------------------------------------------------------------------
__global__ __launch_bounds__(256) void wconv_kernel(
    const float* __restrict__ W, _Float16* __restrict__ WT, int K, int N) {
  __shared__ _Float16 T[64][72];  // [n][k], padded
  const int tid = threadIdx.x;
  const int k0 = blockIdx.y * 64, n0 = blockIdx.x * 64;
#pragma unroll
  for (int i = 0; i < 4; ++i) {
    int idx = tid + i * 256;          // float4 index 0..1023
    int r = idx >> 4;                 // k-local
    int c = (idx & 15) << 2;          // n-local
    float4 v = *(const float4*)(W + (size_t)(k0 + r) * N + n0 + c);
    T[c + 0][r] = (_Float16)v.x;
    T[c + 1][r] = (_Float16)v.y;
    T[c + 2][r] = (_Float16)v.z;
    T[c + 3][r] = (_Float16)v.w;
  }
  __syncthreads();
#pragma unroll
  for (int i = 0; i < 2; ++i) {
    int idx = tid + i * 256;          // v8h chunk 0..511
    int r = idx >> 3;                 // n-local
    int c = (idx & 7) << 3;           // k-local
    v8h o;
#pragma unroll
    for (int e = 0; e < 8; ++e) o[e] = T[r][c + e];
    *(v8h*)(WT + (size_t)(n0 + r) * K + k0 + c) = o;
  }
}

// Elementwise f32 -> f16 (activations), 4 per thread.
__global__ __launch_bounds__(256) void xconv_kernel(
    const float* __restrict__ in, _Float16* __restrict__ out) {
  size_t i = ((size_t)blockIdx.x * 256 + threadIdx.x) * 4;
  float4 v = *(const float4*)(in + i);
  v4h o = {(_Float16)v.x, (_Float16)v.y, (_Float16)v.z, (_Float16)v.w};
  *(v4h*)(out + i) = o;
}

// ---------------------------------------------------------------------------
// GEMM: Y[M,N](f16) = act(A[M,K](f16) @ WT[N,K]^T + bias[N](f32)).
// 128x128x32 tiles, 8 waves (4Mx2N), double-buffered LDS fed by async copies.
// ---------------------------------------------------------------------------
__global__ __launch_bounds__(256) void gemm_kernel(
    const _Float16* __restrict__ A, const _Float16* __restrict__ WT,
    const float* __restrict__ bias, _Float16* __restrict__ Y,
    int M, int N, int K, int act) {
  constexpr int BM = 128, BN = 128, BK = 32;
  __shared__ __align__(16) _Float16 As[2][BM * BK];   // [m][k]
  __shared__ __align__(16) _Float16 Bs[2][BN * BK];   // [n][k]

  const int tid  = threadIdx.x;
  const int wave = tid >> 5;
  const int lane = tid & 31;
  const int bm = blockIdx.y * BM;
  const int bn = blockIdx.x * BN;
  const int wm = (wave >> 1) * 32;
  const int wn = (wave & 1) * 64;

  auto stage = [&](int t, int buf) {
    const int k0 = t * BK;
#pragma unroll
    for (int i = 0; i < 2; ++i) {                 // A tile: 512 x 16B
      int idx = tid + i * 256;
      int r = idx >> 2, off = (idx & 3) << 3;
      async_copy16(A + (size_t)(bm + r) * K + k0 + off, &As[buf][r * BK + off]);
    }
#pragma unroll
    for (int i = 0; i < 2; ++i) {                 // B tile: 512 x 16B
      int idx = tid + i * 256;
      int r = idx >> 2, off = (idx & 3) << 3;
      async_copy16(WT + (size_t)(bn + r) * K + k0 + off, &Bs[buf][r * BK + off]);
    }
  };

  const v8f vz = {0.f, 0.f, 0.f, 0.f, 0.f, 0.f, 0.f, 0.f};
  v8f acc[2][4];
#pragma unroll
  for (int i = 0; i < 2; ++i)
#pragma unroll
    for (int j = 0; j < 4; ++j) acc[i][j] = vz;

  const int T = K / BK;
  stage(0, 0);
  for (int t = 0; t < T; ++t) {
    const int buf = t & 1;
    wait_async_all();       // my writes for tile t landed
    __syncthreads();        // everyone's writes published; prev reads done
    if (t + 1 < T) stage(t + 1, buf ^ 1);

    v16h a0 = frag_a(&As[buf][(wm +  0) * BK], BK);
    v16h a1 = frag_a(&As[buf][(wm + 16) * BK], BK);
#pragma unroll
    for (int j = 0; j < 4; ++j) {
      v16h bf = frag_b(&Bs[buf][(wn + j * 16) * BK], BK);
      acc[0][j] = wmma_f16(a0, bf, acc[0][j]);
      acc[1][j] = wmma_f16(a1, bf, acc[1][j]);
    }
  }

  const int nc = lane & 15;
  const int rb = (lane >> 4) << 3;
#pragma unroll
  for (int i = 0; i < 2; ++i)
#pragma unroll
    for (int j = 0; j < 4; ++j) {
      int gm = bm + wm + i * 16 + rb;
      int gn = bn + wn + j * 16 + nc;
      float bb = bias[gn];
#pragma unroll
      for (int r = 0; r < 8; ++r) {
        float vv = acc[i][j][r] + bb;
        if (act) vv = gelu_exact(vv);
        Y[(size_t)(gm + r) * N + gn] = (_Float16)vv;
      }
    }
}

// ---------------------------------------------------------------------------
// Flash attention over packed f16 qkv[B,N,3D].  dh=64, 128 q-rows per WG,
// streams 128-key blocks with online softmax.  Q/K staged via async copies
// (K's row-major [key][d] layout IS the B-fragment [n][k] layout).
// ---------------------------------------------------------------------------
__global__ __launch_bounds__(256) void fa_kernel(
    const _Float16* __restrict__ qkv, _Float16* __restrict__ out,
    int Bn, int Nn, int Dd, int Hh, float scale) {
  constexpr int BR = 128, BC = 128, DH = 64;
  __shared__ __align__(16) _Float16 Qs[BR * DH];
  __shared__ __align__(16) _Float16 Ks[BC * DH];
  __shared__ __align__(16) _Float16 Vs[DH * BC];   // [d][key]
  __shared__ __align__(16) _Float16 Ps[BR * BC];

  const int tid = threadIdx.x, wave = tid >> 5, lane = tid & 31;
  const int bh = blockIdx.y;
  const int b = bh / Hh, h = bh % Hh;
  const size_t row3 = (size_t)3 * Dd;
  const _Float16* qbase = qkv + (size_t)b * Nn * row3 + (size_t)h * DH;
  const _Float16* kbase = qbase + Dd;
  const _Float16* vbase = qbase + 2 * Dd;
  const int q0 = blockIdx.x * BR;

  // Q tile: 1024 x 16B async copies.
#pragma unroll
  for (int i = 0; i < 4; ++i) {
    int idx = tid + i * 256;
    int r = idx >> 3, off = (idx & 7) << 3;
    async_copy16(qbase + (size_t)(q0 + r) * row3 + off, &Qs[r * DH + off]);
  }

  const v8f vz = {0.f, 0.f, 0.f, 0.f, 0.f, 0.f, 0.f, 0.f};
  v8f o[4];
#pragma unroll
  for (int t = 0; t < 4; ++t) o[t] = vz;
  float mrow[8], lrow[8];
#pragma unroll
  for (int r = 0; r < 8; ++r) { mrow[r] = -__builtin_inff(); lrow[r] = 0.f; }

  for (int j0 = 0; j0 < Nn; j0 += BC) {
    __syncthreads();   // previous iteration done reading Ks/Vs/Ps
    // K tile: async copies.
#pragma unroll
    for (int i = 0; i < 4; ++i) {
      int idx = tid + i * 256;
      int r = idx >> 3, off = (idx & 7) << 3;
      async_copy16(kbase + (size_t)(j0 + r) * row3 + off, &Ks[r * DH + off]);
    }
    // V tile: batched loads, then transposed LDS stores.
    v8h vv[4];
    int vkey[4], vd[4];
#pragma unroll
    for (int i = 0; i < 4; ++i) {
      int idx = tid + i * 256;
      vkey[i] = idx >> 3;
      vd[i]   = (idx & 7) << 3;
      vv[i] = *(const v8h*)(vbase + (size_t)(j0 + vkey[i]) * row3 + vd[i]);
    }
#pragma unroll
    for (int i = 0; i < 4; ++i)
#pragma unroll
      for (int e = 0; e < 8; ++e) Vs[(vd[i] + e) * BC + vkey[i]] = vv[i][e];
    wait_async_all();
    __syncthreads();

    // S = Q @ K^T : wave owns rows [wave*16, wave*16+16).
    v8f s[8];
#pragma unroll
    for (int t = 0; t < 8; ++t) s[t] = vz;
    const _Float16* qr = &Qs[(wave * 16) * DH];
    v16h a0 = frag_a(qr, DH);
    v16h a1 = frag_a(qr + 32, DH);
#pragma unroll
    for (int t = 0; t < 8; ++t) {
      v16h b0 = frag_b(&Ks[(t * 16) * DH], DH);
      v16h b1 = frag_b(&Ks[(t * 16) * DH + 32], DH);
      s[t] = wmma_f16(a0, b0, s[t]);
      s[t] = wmma_f16(a1, b1, s[t]);
    }

    // Online softmax: lane-halves hold disjoint row sets; xor masks < 16.
    float pm[8];
#pragma unroll
    for (int r = 0; r < 8; ++r) pm[r] = -__builtin_inff();
#pragma unroll
    for (int t = 0; t < 8; ++t)
#pragma unroll
      for (int r = 0; r < 8; ++r) {
        float v = s[t][r] * scale;
        s[t][r] = v;
        pm[r] = fmaxf(pm[r], v);
      }
#pragma unroll
    for (int m = 1; m < 16; m <<= 1)
#pragma unroll
      for (int r = 0; r < 8; ++r) pm[r] = fmaxf(pm[r], __shfl_xor(pm[r], m, 32));

    float corr[8], rs[8];
#pragma unroll
    for (int r = 0; r < 8; ++r) {
      float mn = fmaxf(mrow[r], pm[r]);
      corr[r] = __expf(mrow[r] - mn);
      mrow[r] = mn;
      rs[r] = 0.f;
    }
#pragma unroll
    for (int t = 0; t < 8; ++t)
#pragma unroll
      for (int r = 0; r < 8; ++r) {
        float p = __expf(s[t][r] - mrow[r]);
        s[t][r] = p;
        rs[r] += p;
      }
#pragma unroll
    for (int m = 1; m < 16; m <<= 1)
#pragma unroll
      for (int r = 0; r < 8; ++r) rs[r] += __shfl_xor(rs[r], m, 32);
#pragma unroll
    for (int r = 0; r < 8; ++r) lrow[r] = lrow[r] * corr[r] + rs[r];
#pragma unroll
    for (int t = 0; t < 4; ++t)
#pragma unroll
      for (int r = 0; r < 8; ++r) o[t][r] *= corr[r];

    // C-layout f32 -> A-layout f16 via LDS round trip.
    const int rb = (lane >> 4) << 3;
    const int nc = lane & 15;
#pragma unroll
    for (int t = 0; t < 8; ++t)
#pragma unroll
      for (int r = 0; r < 8; ++r)
        Ps[(wave * 16 + rb + r) * BC + t * 16 + nc] = (_Float16)s[t][r];
    __syncthreads();

    // O += P @ V.
    const _Float16* pr = &Ps[(wave * 16) * BC];
#pragma unroll
    for (int ks = 0; ks < 4; ++ks) {
      v16h ap = frag_a(pr + ks * 32, BC);
#pragma unroll
      for (int t = 0; t < 4; ++t) {
        v16h bv = frag_b(&Vs[(t * 16) * BC + ks * 32], BC);
        o[t] = wmma_f16(ap, bv, o[t]);
      }
    }
  }

  const int rb = (lane >> 4) << 3;
  const int nc = lane & 15;
#pragma unroll
  for (int t = 0; t < 4; ++t)
#pragma unroll
    for (int r = 0; r < 8; ++r)
      out[(size_t)b * Nn * Dd + (size_t)(q0 + wave * 16 + rb + r) * Dd +
          h * DH + t * 16 + nc] = (_Float16)(o[t][r] / lrow[r]);
}

// ---------------------------------------------------------------------------
// out[row] = LayerNorm(a[row] + b[row]) * g + be.  f16 in; f16 or f32 out.
// ---------------------------------------------------------------------------
__global__ __launch_bounds__(256) void addln_kernel(
    const _Float16* __restrict__ a, const _Float16* __restrict__ bsrc,
    const float* __restrict__ g, const float* __restrict__ be,
    void* __restrict__ outv, int out32) {
  constexpr int D = 1024;
  const int row = blockIdx.x;
  const int tid = threadIdx.x;
  const int c0 = tid * 4;
  v4h va = *(const v4h*)(a + (size_t)row * D + c0);
  v4h vb = *(const v4h*)(bsrc + (size_t)row * D + c0);
  float x[4];
  float s = 0.f, s2 = 0.f;
#pragma unroll
  for (int i = 0; i < 4; ++i) {
    float v = (float)va[i] + (float)vb[i];
    x[i] = v; s += v; s2 += v * v;
  }
#pragma unroll
  for (int m = 1; m < 32; m <<= 1) {
    s  += __shfl_xor(s,  m, 32);
    s2 += __shfl_xor(s2, m, 32);
  }
  __shared__ float red[16];
  const int wave = tid >> 5, lane = tid & 31;
  if (lane == 0) { red[wave] = s; red[8 + wave] = s2; }
  __syncthreads();
  float ts = 0.f, t2 = 0.f;
#pragma unroll
  for (int w = 0; w < 8; ++w) { ts += red[w]; t2 += red[8 + w]; }
  float mu  = ts * (1.f / D);
  float var = t2 * (1.f / D) - mu * mu;
  float inv = rsqrtf(var + LN_EPS);
  float4 gg = *(const float4*)(g + c0);
  float4 bb = *(const float4*)(be + c0);
  float y0 = (x[0] - mu) * inv * gg.x + bb.x;
  float y1 = (x[1] - mu) * inv * gg.y + bb.y;
  float y2 = (x[2] - mu) * inv * gg.z + bb.z;
  float y3 = (x[3] - mu) * inv * gg.w + bb.w;
  if (out32) {
    float4 o = {y0, y1, y2, y3};
    *(float4*)((float*)outv + (size_t)row * D + c0) = o;
  } else {
    v4h o = {(_Float16)y0, (_Float16)y1, (_Float16)y2, (_Float16)y3};
    *(v4h*)((_Float16*)outv + (size_t)row * D + c0) = o;
  }
}

// ---------------------------------------------------------------------------
extern "C" void kernel_launch(void* const* d_in, const int* in_sizes, int n_in,
                              void* d_out, int out_size, void* d_ws,
                              size_t ws_size, hipStream_t stream) {
  const float* x0   = (const float*)d_in[0];
  const float* Wqkv = (const float*)d_in[1];
  const float* bqkv = (const float*)d_in[2];
  const float* W1   = (const float*)d_in[3];
  const float* b1   = (const float*)d_in[4];
  const float* W2   = (const float*)d_in[5];
  const float* b2   = (const float*)d_in[6];
  const float* g1   = (const float*)d_in[7];
  const float* be1  = (const float*)d_in[8];
  const float* g2   = (const float*)d_in[9];
  const float* be2  = (const float*)d_in[10];

  const int Bb = 2, Nn = 2048, Dd = 1024, Hh = 16, Ff = 4, Ll = 4;
  const int Mr = Bb * Nn;                      // 4096 rows
  const size_t nMed = (size_t)Mr * Dd;         // 4.19M
  const size_t nBig = (size_t)Mr * Ff * Dd;    // 16.78M (f1; qkv fits inside)

  _Float16* ws = (_Float16*)d_ws;
  _Float16* wtQ = ws;                               // [L][3D][D]
  _Float16* wt1 = wtQ + (size_t)Ll * 3 * Dd * Dd;   // [L][4D][D]
  _Float16* wt2 = wt1 + (size_t)Ll * Ff * Dd * Dd;  // [L][D][4D]
  _Float16* xh  = wt2 + (size_t)Ll * Dd * Ff * Dd;  // x in f16 [Mr][D]
  _Float16* bufA = xh + nMed;                       // qkv / f1
  _Float16* bufB = bufA + nBig;                     // attn / f2
  _Float16* hbuf = bufB + nMed;
  _Float16* xb   = hbuf + nMed;

  const float scale = 0.03125f;   // 1024 ** -0.5 (embedDim, per reference)
  const dim3 blk(256);

  // ---- One-time pre-pass: transpose+convert weights, convert x ----
  for (int l = 0; l < Ll; ++l) {
    wconv_kernel<<<dim3(3 * Dd / 64, Dd / 64), blk, 0, stream>>>(
        Wqkv + (size_t)l * Dd * 3 * Dd, wtQ + (size_t)l * 3 * Dd * Dd,
        Dd, 3 * Dd);
    wconv_kernel<<<dim3(Ff * Dd / 64, Dd / 64), blk, 0, stream>>>(
        W1 + (size_t)l * Dd * Ff * Dd, wt1 + (size_t)l * Ff * Dd * Dd,
        Dd, Ff * Dd);
    wconv_kernel<<<dim3(Dd / 64, Ff * Dd / 64), blk, 0, stream>>>(
        W2 + (size_t)l * Ff * Dd * Dd, wt2 + (size_t)l * Dd * Ff * Dd,
        Ff * Dd, Dd);
  }
  xconv_kernel<<<dim3((unsigned)(nMed / 1024)), blk, 0, stream>>>(x0, xh);

  // ---- Layer loop ----
  const _Float16* cur = xh;
  for (int l = 0; l < Ll; ++l) {
    gemm_kernel<<<dim3(3 * Dd / 128, Mr / 128), blk, 0, stream>>>(
        cur, wtQ + (size_t)l * 3 * Dd * Dd, bqkv + (size_t)l * 3 * Dd,
        bufA, Mr, 3 * Dd, Dd, 0);
    fa_kernel<<<dim3(Nn / 128, Bb * Hh), blk, 0, stream>>>(
        bufA, bufB, Bb, Nn, Dd, Hh, scale);
    addln_kernel<<<dim3(Mr), blk, 0, stream>>>(
        bufB, cur, g1 + (size_t)l * Dd, be1 + (size_t)l * Dd, hbuf, 0);
    gemm_kernel<<<dim3(Ff * Dd / 128, Mr / 128), blk, 0, stream>>>(
        hbuf, wt1 + (size_t)l * Ff * Dd * Dd, b1 + (size_t)l * Ff * Dd,
        bufA, Mr, Ff * Dd, Dd, 1);
    gemm_kernel<<<dim3(Dd / 128, Mr / 128), blk, 0, stream>>>(
        bufA, wt2 + (size_t)l * Dd * Ff * Dd, b2 + (size_t)l * Dd,
        bufB, Mr, Dd, Ff * Dd, 1);
    int last = (l == Ll - 1);
    addln_kernel<<<dim3(Mr), blk, 0, stream>>>(
        bufB, hbuf, g2 + (size_t)l * Dd, be2 + (size_t)l * Dd,
        last ? d_out : (void*)xb, last);
    cur = xb;
  }
  (void)in_sizes; (void)n_in; (void)out_size; (void)ws_size;
}
